// Seq2SeqWithAttention_71983651881144
// MI455X (gfx1250) — matrix operand order repaired and checked by reading, not compiled, not run on hardware
//
#include <hip/hip_runtime.h>

#define DEVINL __device__ __forceinline__

typedef _Float16 half16 __attribute__((ext_vector_type(16)));
typedef float    f32x8  __attribute__((ext_vector_type(8)));
typedef unsigned int u32x4 __attribute__((ext_vector_type(4)));

union Frag { half16 v; u32x4 q[2]; };

DEVINL float sigf(float x)      { return 1.0f / (1.0f + __expf(-x)); }
DEVINL float tanhfast(float x)  { return 1.0f - 2.0f / (__expf(2.0f * x) + 1.0f); }

// Swizzled B layout: for logical B[K,N], element (k,n) lives at
// (((k/32)*tilesN + n/16)*32 + lane)*16 + (k%8) + ((k/16)&1)*8, lane = ((k/8)&1)*16 + (n%16).
// => each lane of a 32x16 fragment owns 16 contiguous f16 (two b128 loads).
DEVINL size_t swz(int k, int n, int tilesN) {
  int kt = k >> 5, nt = n >> 4;
  int lane = (((k >> 3) & 1) << 4) | (n & 15);
  int j = (k & 7) | (((k >> 4) & 1) << 3);
  return (((size_t)kt * tilesN + nt) * 32 + lane) * (size_t)16 + j;
}

// ---------------- utility kernels ----------------

__global__ void k_addvec(const float* __restrict__ a, const float* __restrict__ b,
                         float* __restrict__ o, int n) {
  int i = blockIdx.x * blockDim.x + threadIdx.x;
  if (i < n) o[i] = a[i] + b[i];
}

// W: [N,K] fp32 row-major (logical B[k][n] = W[n][k]); convert+swizzle to f16.
__global__ void k_wswz(const float* __restrict__ W, _Float16* __restrict__ Bsw,
                       int N, int K) {
  size_t i = (size_t)blockIdx.x * blockDim.x + threadIdx.x;
  if (i >= (size_t)N * K) return;
  int n = (int)(i / K), k = (int)(i % K);
  Bsw[swz(k, n, N >> 4)] = (_Float16)W[i];
}

// trg embedding gather into [T*B, E] row-major f16 (row r = t*B + b)
__global__ void k_embed_trg(const int* __restrict__ trg, const float* __restrict__ tab,
                            _Float16* __restrict__ out, int TB, int E, int Bsz, int T) {
  size_t i = (size_t)blockIdx.x * blockDim.x + threadIdx.x;
  if (i >= (size_t)TB * E) return;
  int r = (int)(i / E), e = (int)(i % E);
  int tok = trg[(r % Bsz) * T + r / Bsz];
  out[i] = (_Float16)tab[(size_t)tok * E + e];
}

// src embedding gather, written twice in swizzled B layouts:
//   scores B: K=E, N=S (i.e. src_emb transposed per batch)
//   context B: K=S, N=E
__global__ void k_embed_src(const int* __restrict__ src, const float* __restrict__ tab,
                            _Float16* __restrict__ BswScr, _Float16* __restrict__ BswCtx,
                            int Bsz, int S, int E) {
  size_t i = (size_t)blockIdx.x * blockDim.x + threadIdx.x;
  if (i >= (size_t)Bsz * S * E) return;
  int e = (int)(i % E);
  int s = (int)((i / E) % S);
  int b = (int)(i / ((size_t)E * S));
  _Float16 v = (_Float16)tab[(size_t)src[b * S + s] * E + e];
  size_t base = (size_t)b * S * E;
  BswScr[base + swz(e, s, S >> 4)] = v;
  BswCtx[base + swz(s, e, E >> 4)] = v;
}

__global__ void k_zero_state(float* __restrict__ c, _Float16* __restrict__ h, int n) {
  int i = blockIdx.x * blockDim.x + threadIdx.x;
  if (i < n) { c[i] = 0.0f; h[i] = (_Float16)0.0f; }
}

// ------- small WMMA GEMM (1 tile/wave) for batched attention GEMMs -------
template<bool BIAS, bool TANH, bool OUTF32, bool OUTF16, bool REMAP>
__global__ void k_gemm(const _Float16* __restrict__ A, long sA, int lda,
                       const _Float16* __restrict__ Bsw, long sB,
                       const float* __restrict__ bias,
                       float* __restrict__ D32, long sD32, int ldd32,
                       _Float16* __restrict__ D16, long sD16, int ldd16, int coff,
                       int tilesM, int tilesN, int K, float scale, int RD, int RT) {
  const int z = blockIdx.y;
  A += (size_t)z * sA;
  Bsw += (size_t)z * sB;
  const int wid = blockIdx.x * (blockDim.x >> 5) + (threadIdx.x >> 5);
  if (wid >= tilesM * tilesN) return;
  const int tM = wid / tilesN, tN = wid % tilesN;
  const int lane = threadIdx.x & 31;
  const int hf = lane >> 4;

  const _Float16* ap = A + (size_t)(tM * 16 + (lane & 15)) * lda + hf * 8;
  const _Float16* bp = Bsw + ((size_t)tN * 32 + lane) * 16;
  const size_t bstep = (size_t)tilesN * 512;

  f32x8 acc = {0.f, 0.f, 0.f, 0.f, 0.f, 0.f, 0.f, 0.f};
  for (int k0 = 0; k0 < K; k0 += 32) {
    Frag a, b;
    a.q[0] = *(const u32x4*)(ap + k0);
    a.q[1] = *(const u32x4*)(ap + k0 + 16);
    b.q[0] = *(const u32x4*)(bp);
    b.q[1] = *(const u32x4*)(bp + 8);
    bp += bstep;
    acc = __builtin_amdgcn_wmma_f32_16x16x32_f16(false, a.v, false, b.v,
                                                 (short)0, acc, false, false);
  }

  const int n = tN * 16 + (lane & 15);
  const float bval = BIAS ? bias[n] : 0.0f;
#pragma unroll
  for (int v = 0; v < 8; ++v) {
    int row = tM * 16 + v + 8 * hf;
    float val = acc[v] * scale + bval;
    if (TANH) val = tanhfast(val);
    if (OUTF32) D32[(size_t)z * sD32 + (size_t)row * ldd32 + n] = val;
    if (OUTF16) {
      int r2 = REMAP ? (row % RD) * RT + row / RD : row;
      D16[(size_t)z * sD16 + (size_t)r2 * ldd16 + coff + n] = (_Float16)val;
    }
  }
}

// ------- big WMMA GEMM: MT x NT register tiles per wave -------
// 4x4: 16 accumulators (128 VGPRs); 8 fragment loads feed 16 WMMAs per K-step,
// cutting L2 read traffic 4x vs 1 tile/wave. STREAM=nontemporal fp32 stores.
template<int MT, int NT, bool BIAS, bool TANH, bool OUTF32, bool OUTF16,
         bool REMAP, bool STREAM>
__global__ void k_gemm_big(const _Float16* __restrict__ A, int lda,
                           const _Float16* __restrict__ Bsw,
                           const float* __restrict__ bias,
                           float* __restrict__ D32, int ldd32,
                           _Float16* __restrict__ D16, int ldd16, int coff,
                           int tilesM, int tilesN, int K, float scale,
                           int RD, int RT) {
  const int tilesM2 = tilesM / MT, tilesN2 = tilesN / NT;
  const int wid = blockIdx.x * (blockDim.x >> 5) + (threadIdx.x >> 5);
  if (wid >= tilesM2 * tilesN2) return;
  const int tM2 = wid / tilesN2, tN2 = wid % tilesN2;
  const int lane = threadIdx.x & 31;
  const int hf = lane >> 4;

  const _Float16* ap = A + (size_t)(tM2 * MT * 16 + (lane & 15)) * lda + hf * 8;
  const _Float16* bp = Bsw + (((size_t)tN2 * NT) * 32 + lane) * 16;
  const size_t bstep = (size_t)tilesN * 512;   // full tile-row stride per K-step
  const size_t arow = (size_t)16 * lda;        // A stride per M-tile

  f32x8 acc[MT][NT];
#pragma unroll
  for (int mt = 0; mt < MT; ++mt)
#pragma unroll
    for (int nt = 0; nt < NT; ++nt)
      acc[mt][nt] = f32x8{0.f, 0.f, 0.f, 0.f, 0.f, 0.f, 0.f, 0.f};

  for (int k0 = 0; k0 < K; k0 += 32) {
    Frag af[MT], bf[NT];
#pragma unroll
    for (int mt = 0; mt < MT; ++mt) {
      af[mt].q[0] = *(const u32x4*)(ap + mt * arow + k0);
      af[mt].q[1] = *(const u32x4*)(ap + mt * arow + k0 + 16);
    }
#pragma unroll
    for (int nt = 0; nt < NT; ++nt) {
      bf[nt].q[0] = *(const u32x4*)(bp + nt * 512);
      bf[nt].q[1] = *(const u32x4*)(bp + nt * 512 + 8);
    }
    bp += bstep;
#pragma unroll
    for (int mt = 0; mt < MT; ++mt)
#pragma unroll
      for (int nt = 0; nt < NT; ++nt)
        acc[mt][nt] = __builtin_amdgcn_wmma_f32_16x16x32_f16(
            false, af[mt].v, false, bf[nt].v, (short)0, acc[mt][nt], false, false);
  }

#pragma unroll
  for (int nt = 0; nt < NT; ++nt) {
    const int n = (tN2 * NT + nt) * 16 + (lane & 15);
    const float bval = BIAS ? bias[n] : 0.0f;
#pragma unroll
    for (int mt = 0; mt < MT; ++mt) {
#pragma unroll
      for (int v = 0; v < 8; ++v) {
        int row = (tM2 * MT + mt) * 16 + v + 8 * hf;
        float val = acc[mt][nt][v] * scale + bval;
        if (TANH) val = tanhfast(val);
        if (OUTF32) {
          float* p = D32 + (size_t)row * ldd32 + n;
          if (STREAM) __builtin_nontemporal_store(val, p);
          else *p = val;
        }
        if (OUTF16) {
          int r2 = REMAP ? (row % RD) * RT + row / RD : row;
          D16[(size_t)r2 * ldd16 + coff + n] = (_Float16)val;
        }
      }
    }
  }
}

// ---------------- LSTM recurrent step ----------------
__global__ void k_lstm_step(const float* __restrict__ Xt,        // [B,4H]
                            const _Float16* __restrict__ hprev,  // [B,H]
                            const _Float16* __restrict__ WhhSw,  // swizzled K=H,N=4H
                            float* __restrict__ cstate,          // [B,H]
                            _Float16* __restrict__ hout) {       // [B,H]
  const int H = 512;
  const int tM = blockIdx.x;
  const int tN = blockIdx.y;
  const int lane = threadIdx.x & 31;
  const int hf = lane >> 4;
  const int tilesN = 128;  // 4H/16

  const _Float16* ap = hprev + (size_t)(tM * 16 + (lane & 15)) * H + hf * 8;
  const size_t laneoff = ((size_t)tN * 32 + lane) * 16;
  const size_t goff = (size_t)(H >> 4) * 512;
  const size_t ktstep = (size_t)tilesN * 512;

  f32x8 aI = {0.f,0.f,0.f,0.f,0.f,0.f,0.f,0.f};
  f32x8 aF = aI, aG = aI, aO = aI;

  for (int k0 = 0; k0 < H; k0 += 32) {
    Frag a;
    a.q[0] = *(const u32x4*)(ap + k0);
    a.q[1] = *(const u32x4*)(ap + k0 + 16);
    const _Float16* bb = WhhSw + (size_t)(k0 >> 5) * ktstep + laneoff;
    Frag b0, b1, b2, b3;
    b0.q[0] = *(const u32x4*)(bb);              b0.q[1] = *(const u32x4*)(bb + 8);
    b1.q[0] = *(const u32x4*)(bb + goff);       b1.q[1] = *(const u32x4*)(bb + goff + 8);
    b2.q[0] = *(const u32x4*)(bb + 2 * goff);   b2.q[1] = *(const u32x4*)(bb + 2 * goff + 8);
    b3.q[0] = *(const u32x4*)(bb + 3 * goff);   b3.q[1] = *(const u32x4*)(bb + 3 * goff + 8);
    aI = __builtin_amdgcn_wmma_f32_16x16x32_f16(false, a.v, false, b0.v, (short)0, aI, false, false);
    aF = __builtin_amdgcn_wmma_f32_16x16x32_f16(false, a.v, false, b1.v, (short)0, aF, false, false);
    aG = __builtin_amdgcn_wmma_f32_16x16x32_f16(false, a.v, false, b2.v, (short)0, aG, false, false);
    aO = __builtin_amdgcn_wmma_f32_16x16x32_f16(false, a.v, false, b3.v, (short)0, aO, false, false);
  }

  const int n = tN * 16 + (lane & 15);
#pragma unroll
  for (int v = 0; v < 8; ++v) {
    int brow = tM * 16 + v + 8 * hf;
    const float* xr = Xt + (size_t)brow * (4 * H);
    float gi = sigf(xr[n] + aI[v]);
    float gf = sigf(xr[H + n] + aF[v]);
    float gg = tanhfast(xr[2 * H + n] + aG[v]);
    float go = sigf(xr[3 * H + n] + aO[v]);
    size_t ci = (size_t)brow * H + n;
    float cn = gf * cstate[ci] + gi * gg;
    cstate[ci] = cn;
    hout[ci] = (_Float16)(go * tanhfast(cn));
  }
}

// ---------------- softmax over S=128 ----------------
__global__ void k_softmax(const float* __restrict__ scores, float* __restrict__ attn,
                          _Float16* __restrict__ attn16, int S) {
  int row = blockIdx.x, t = threadIdx.x;
  float x = scores[(size_t)row * S + t];
  __shared__ float red[128];
  red[t] = x; __syncthreads();
  for (int o = 64; o > 0; o >>= 1) { if (t < o) red[t] = fmaxf(red[t], red[t + o]); __syncthreads(); }
  float mx = red[0]; __syncthreads();
  float e = __expf(x - mx);
  red[t] = e; __syncthreads();
  for (int o = 64; o > 0; o >>= 1) { if (t < o) red[t] += red[t + o]; __syncthreads(); }
  float p = e / red[0];
  attn[(size_t)row * S + t] = p;
  attn16[(size_t)row * S + t] = (_Float16)p;
}

// ---------------- host orchestration ----------------
extern "C" void kernel_launch(void* const* d_in, const int* in_sizes, int n_in,
                              void* d_out, int out_size, void* d_ws, size_t ws_size,
                              hipStream_t stream) {
  (void)in_sizes; (void)n_in; (void)out_size; (void)ws_size;
  const int Bz = 32, S = 128, T = 128, V = 32000, E = 512, H = 512;
  const int TB = T * Bz;  // 4096

  const int*   src     = (const int*)d_in[0];
  const int*   trg     = (const int*)d_in[1];
  const float* src_tab = (const float*)d_in[2];
  const float* dec_tab = (const float*)d_in[3];
  const float* W_ih0   = (const float*)d_in[4];
  const float* W_hh0   = (const float*)d_in[5];
  const float* b_ih0   = (const float*)d_in[6];
  const float* b_hh0   = (const float*)d_in[7];
  const float* W_ih1   = (const float*)d_in[8];
  const float* W_hh1   = (const float*)d_in[9];
  const float* b_ih1   = (const float*)d_in[10];
  const float* b_hh1   = (const float*)d_in[11];
  const float* fc_W    = (const float*)d_in[12];
  const float* fc_b    = (const float*)d_in[13];
  const float* h2o_W   = (const float*)d_in[14];
  const float* h2o_b   = (const float*)d_in[15];
  const float* out_b   = (const float*)d_in[16];

  float* logits = (float*)d_out;
  float* attn   = logits + (size_t)Bz * T * V;

  char* base = (char*)d_ws; size_t off = 0;
  auto alloc = [&](size_t bytes) -> void* {
    void* p = base + off; off = (off + bytes + 255) & ~(size_t)255; return p;
  };
  float*    bias01 = (float*)alloc((size_t)4 * H * 4);
  float*    bias11 = (float*)alloc((size_t)4 * H * 4);
  _Float16* BswIh0 = (_Float16*)alloc((size_t)E * 4 * H * 2);
  _Float16* BswHh0 = (_Float16*)alloc((size_t)H * 4 * H * 2);
  _Float16* BswIh1 = (_Float16*)alloc((size_t)H * 4 * H * 2);
  _Float16* BswHh1 = (_Float16*)alloc((size_t)H * 4 * H * 2);
  _Float16* BswFc  = (_Float16*)alloc((size_t)H * H * 2);
  _Float16* BswH2o = (_Float16*)alloc((size_t)(H + 512) * E * 2);
  _Float16* BswEmb = (_Float16*)alloc((size_t)E * V * 2);
  _Float16* trgEmb = (_Float16*)alloc((size_t)TB * E * 2);
  _Float16* BswScr = (_Float16*)alloc((size_t)Bz * E * S * 2);
  _Float16* BswCtx = (_Float16*)alloc((size_t)Bz * S * E * 2);
  float*    X      = (float*)alloc((size_t)TB * 4 * H * 4);   // reused for both layers
  _Float16* h1seq  = (_Float16*)alloc((size_t)TB * H * 2);
  _Float16* h2seq  = (_Float16*)alloc((size_t)TB * H * 2);
  _Float16* hzero  = (_Float16*)alloc((size_t)Bz * H * 2);
  float*    cbuf   = (float*)alloc((size_t)Bz * H * 4);
  _Float16* concat = (_Float16*)alloc((size_t)TB * (H + 512) * 2);
  float*    scrB   = (float*)alloc((size_t)Bz * T * S * 4);
  _Float16* attn16 = (_Float16*)alloc((size_t)Bz * T * S * 2);
  _Float16* pre16  = (_Float16*)alloc((size_t)TB * E * 2);

  auto cdiv = [](long a, long b) { return (int)((a + b - 1) / b); };

  k_addvec<<<cdiv(4 * H, 256), 256, 0, stream>>>(b_ih0, b_hh0, bias01, 4 * H);
  k_addvec<<<cdiv(4 * H, 256), 256, 0, stream>>>(b_ih1, b_hh1, bias11, 4 * H);

  k_wswz<<<cdiv((long)4 * H * E, 256), 256, 0, stream>>>(W_ih0, BswIh0, 4 * H, E);
  k_wswz<<<cdiv((long)4 * H * H, 256), 256, 0, stream>>>(W_hh0, BswHh0, 4 * H, H);
  k_wswz<<<cdiv((long)4 * H * H, 256), 256, 0, stream>>>(W_ih1, BswIh1, 4 * H, H);
  k_wswz<<<cdiv((long)4 * H * H, 256), 256, 0, stream>>>(W_hh1, BswHh1, 4 * H, H);
  k_wswz<<<cdiv((long)H * H, 256), 256, 0, stream>>>(fc_W, BswFc, H, H);
  k_wswz<<<cdiv((long)E * (H + 512), 256), 256, 0, stream>>>(h2o_W, BswH2o, E, H + 512);
  k_wswz<<<cdiv((long)V * E, 256), 256, 0, stream>>>(dec_tab, BswEmb, V, E);

  k_embed_trg<<<cdiv((long)TB * E, 256), 256, 0, stream>>>(trg, dec_tab, trgEmb, TB, E, Bz, T);
  k_embed_src<<<cdiv((long)Bz * S * E, 256), 256, 0, stream>>>(src, src_tab, BswScr, BswCtx, Bz, S, E);

  // X0 = trgEmb @ W_ih0^T + (b_ih0 + b_hh0)   [4096 x 2048 x 512]
  k_gemm_big<4, 4, true, false, true, false, false, false>
      <<<dim3(cdiv((long)(TB / 64) * (4 * H / 64), 4), 1), 128, 0, stream>>>(
      trgEmb, E, BswIh0, bias01,
      X, 4 * H, (_Float16*)nullptr, 0, 0,
      TB / 16, 4 * H / 16, E, 1.0f, 0, 0);

  // decoder LSTM layer 0 (sequential over T)
  k_zero_state<<<cdiv((long)Bz * H, 256), 256, 0, stream>>>(cbuf, hzero, Bz * H);
  for (int t = 0; t < T; ++t) {
    const _Float16* hp = (t == 0) ? hzero : (h1seq + (size_t)(t - 1) * Bz * H);
    k_lstm_step<<<dim3(Bz / 16, H / 16), 32, 0, stream>>>(
        X + (size_t)t * Bz * 4 * H, hp, BswHh0, cbuf, h1seq + (size_t)t * Bz * H);
  }

  // X1 = h1seq @ W_ih1^T + (b_ih1 + b_hh1)  (reuse X)
  k_gemm_big<4, 4, true, false, true, false, false, false>
      <<<dim3(cdiv((long)(TB / 64) * (4 * H / 64), 4), 1), 128, 0, stream>>>(
      h1seq, H, BswIh1, bias11,
      X, 4 * H, (_Float16*)nullptr, 0, 0,
      TB / 16, 4 * H / 16, H, 1.0f, 0, 0);

  // decoder LSTM layer 1
  k_zero_state<<<cdiv((long)Bz * H, 256), 256, 0, stream>>>(cbuf, hzero, Bz * H);
  for (int t = 0; t < T; ++t) {
    const _Float16* hp = (t == 0) ? hzero : (h2seq + (size_t)(t - 1) * Bz * H);
    k_lstm_step<<<dim3(Bz / 16, H / 16), 32, 0, stream>>>(
        X + (size_t)t * Bz * 4 * H, hp, BswHh1, cbuf, h2seq + (size_t)t * Bz * H);
  }

  // fc_out -> concat[:, 0:512], remap rows (t,b)->(b,t)
  k_gemm_big<4, 4, true, false, false, true, true, false>
      <<<dim3(cdiv((long)(TB / 64) * (H / 64), 4), 1), 128, 0, stream>>>(
      h2seq, H, BswFc, fc_b,
      (float*)nullptr, 0, concat, H + 512, 0,
      TB / 16, H / 16, H, 1.0f, Bz, T);

  // attention scores (batched over b): out @ src_emb^T / sqrt(512)
  k_gemm<false, false, true, false, false>
      <<<dim3(cdiv((long)(T / 16) * (S / 16), 4), Bz), 128, 0, stream>>>(
      concat, (long)T * (H + 512), H + 512, BswScr, (long)E * S, (const float*)nullptr,
      scrB, (long)T * S, S, (_Float16*)nullptr, 0, 0, 0,
      T / 16, S / 16, H, 0.04419417382415922f, 0, 0);

  // softmax rows [B*T, S] -> attn (fp32 output) + f16 copy for context GEMM
  k_softmax<<<Bz * T, S, 0, stream>>>(scrB, attn, attn16, S);

  // context (batched): attn @ src_emb -> concat[:, 512:1024]
  k_gemm<false, false, false, true, false>
      <<<dim3(cdiv((long)(T / 16) * (E / 16), 4), Bz), 128, 0, stream>>>(
      attn16, (long)T * S, S, BswCtx, (long)S * E, (const float*)nullptr,
      (float*)nullptr, 0, 0, concat, (long)T * (H + 512), H + 512, H,
      T / 16, E / 16, S, 1.0f, 0, 0);

  // pre = tanh(concat @ h2o_W^T + h2o_b)
  k_gemm_big<4, 4, true, true, false, true, false, false>
      <<<dim3(cdiv((long)(TB / 64) * (E / 64), 4), 1), 128, 0, stream>>>(
      concat, H + 512, BswH2o, h2o_b,
      (float*)nullptr, 0, pre16, E, 0,
      TB / 16, E / 16, H + 512, 1.0f, 0, 0);

  // logits = pre @ dec_emb^T + out_b  (dominant GEMM; NT stores keep BswEmb in L2)
  k_gemm_big<4, 4, true, false, true, false, false, true>
      <<<dim3(cdiv((long)(TB / 64) * (V / 64), 4), 1), 128, 0, stream>>>(
      pre16, E, BswEmb, out_b,
      logits, V, (_Float16*)nullptr, 0, 0,
      TB / 16, V / 16, E, 1.0f, 0, 0);
}
